// GCN_61598420959671
// MI455X (gfx1250) — compile-verified, matrix operand order checked by reading, not run on hardware
//
#include <hip/hip_runtime.h>
#include <hip/hip_bf16.h>

// ---------------------------------------------------------------------------
// GCN (3x GCNConv + FC) for MI455X (gfx1250, wave32, WMMA)
//   - dense transforms: v_wmma_f32_16x16x32_bf16, activations stored bf16
//   - scatter-add: fp32 global atomics (accuracy), bf16 gathers (bandwidth)
// ---------------------------------------------------------------------------

#define NUM_NODES 64
#define BATCH     8192
#define NTOT      (NUM_NODES * BATCH)   // 524288
#define F_IN      16
#define H1        256
#define H2        512
#define H3        128
#define NEDGE     (4 * NTOT)            // 2097152
#define SLOPE     0.01f

typedef __bf16 v8bf  __attribute__((ext_vector_type(8)));
typedef __bf16 v16bf __attribute__((ext_vector_type(16)));
typedef float  v8f   __attribute__((ext_vector_type(8)));

using bf16_t = __hip_bfloat16;

// ---------------------------------------------------------------------------
// WMMA GEMM:  C[M,Nc] (bf16) = A[M,K] (bf16, row-major, ld=K)
//                             x Bt[Nc,K]^T (bf16, weights pre-transposed)
// Block: 4 waves (128 threads). Block tile 64x64, each wave does 16x64.
// A fragment layout (16-bit A 16x32): lanes 0-15 -> K {0..7,16..23},
// lanes 16-31 -> K {8..15,24..31}; B symmetric with N = lane&15.
// ---------------------------------------------------------------------------
__global__ void __launch_bounds__(128)
wmma_gemm_bf16(const bf16_t* __restrict__ A,
               const bf16_t* __restrict__ Bt,
               bf16_t* __restrict__ C,
               int M, int K, int Nc)
{
    const int lane    = threadIdx.x & 31;
    const int wave    = threadIdx.x >> 5;
    const int halfsel = lane >> 4;          // 0 or 1
    const int l16     = lane & 15;
    const int khalf   = halfsel << 3;       // 0 or 8

    const int rowBase = blockIdx.x * 64 + wave * 16;
    const int colBase = blockIdx.y * 64;

    v8f acc[4] = {};

    const bf16_t* aPtr = A + (size_t)(rowBase + l16) * K + khalf;

    for (int k0 = 0; k0 < K; k0 += 32) {
        v8bf alo = *reinterpret_cast<const v8bf*>(aPtr + k0);
        v8bf ahi = *reinterpret_cast<const v8bf*>(aPtr + k0 + 16);
        v16bf a = __builtin_shufflevector(alo, ahi,
                    0,1,2,3,4,5,6,7,8,9,10,11,12,13,14,15);
#pragma unroll
        for (int t = 0; t < 4; ++t) {
            const bf16_t* bPtr =
                Bt + (size_t)(colBase + t * 16 + l16) * K + k0 + khalf;
            v8bf blo = *reinterpret_cast<const v8bf*>(bPtr);
            v8bf bhi = *reinterpret_cast<const v8bf*>(bPtr + 16);
            v16bf b = __builtin_shufflevector(blo, bhi,
                        0,1,2,3,4,5,6,7,8,9,10,11,12,13,14,15);
            acc[t] = __builtin_amdgcn_wmma_f32_16x16x32_bf16(
                         false, a, false, b, (short)0, acc[t], false, false);
        }
    }

    // C/D layout: VGPR v -> M = v + 8*halfsel, N = lane&15
#pragma unroll
    for (int t = 0; t < 4; ++t) {
        int col = colBase + t * 16 + l16;
#pragma unroll
        for (int v = 0; v < 8; ++v) {
            int row = rowBase + (halfsel << 3) + v;
            C[(size_t)row * Nc + col] = __float2bfloat16(acc[t][v]);
        }
    }
}

// ---------------------------------------------------------------------------
// Degree / normalization
// ---------------------------------------------------------------------------
__global__ void deg_init_kernel(float* __restrict__ deg, int n)
{
    int i = blockIdx.x * blockDim.x + threadIdx.x;
    if (i < n) deg[i] = 1.0f;               // self-loop contribution
}

__global__ void deg_acc_kernel(const int* __restrict__ dst,
                               float* __restrict__ deg, int e)
{
    int i = blockIdx.x * blockDim.x + threadIdx.x;
    if (i < e) atomicAdd(&deg[dst[i]], 1.0f);
}

__global__ void deg_rsqrt_kernel(float* __restrict__ deg, int n)
{
    int i = blockIdx.x * blockDim.x + threadIdx.x;
    if (i < n) deg[i] = rsqrtf(deg[i]);     // deg >= 1 always
}

// ---------------------------------------------------------------------------
// Conversions: fp32 -> bf16 (input, padded K) and weight transpose
// ---------------------------------------------------------------------------
__global__ void conv_x_kernel(const float* __restrict__ x,
                              bf16_t* __restrict__ xb)
{
    int t = blockIdx.x * blockDim.x + threadIdx.x;
    if (t >= NTOT * 32) return;
    int node = t >> 5, k = t & 31;          // pad K 16 -> 32 with zeros
    xb[t] = __float2bfloat16(k < F_IN ? x[node * F_IN + k] : 0.0f);
}

__global__ void conv_wt_kernel(const float* __restrict__ Wsrc,
                               bf16_t* __restrict__ Wt,
                               int K, int Nc, int Kpad)
{
    int t = blockIdx.x * blockDim.x + threadIdx.x;
    if (t >= Nc * Kpad) return;
    int n = t / Kpad, k = t % Kpad;
    Wt[t] = __float2bfloat16(k < K ? Wsrc[k * Nc + n] : 0.0f);
}

// ---------------------------------------------------------------------------
// Scatter phase (lc = log2(channels), channels power of 2)
// ---------------------------------------------------------------------------
__global__ void self_loop_init_kernel(const bf16_t* __restrict__ h,
                                      const float* __restrict__ dinv,
                                      float* __restrict__ acc,
                                      int lc, long long total)
{
    long long t = (long long)blockIdx.x * blockDim.x + threadIdx.x;
    if (t >= total) return;
    int node = (int)(t >> lc);
    float di = dinv[node];
    acc[t] = __bfloat162float(h[t]) * di * di;
}

__global__ void edge_scatter_kernel(const bf16_t* __restrict__ h,
                                    const float* __restrict__ dinv,
                                    const int* __restrict__ src,
                                    const int* __restrict__ dst,
                                    float* __restrict__ acc,
                                    int lc, long long total)
{
    long long t = (long long)blockIdx.x * blockDim.x + threadIdx.x;
    if (t >= total) return;
    int e = (int)(t >> lc);
    int c = (int)(t & ((1 << lc) - 1));
    int s = src[e], d = dst[e];
    float w = dinv[s] * dinv[d];
    float m = w * __bfloat162float(h[((size_t)s << lc) + c]);
    atomicAdd(acc + ((size_t)d << lc) + c, m);
}

__global__ void bias_act_kernel(const float* __restrict__ acc,
                                const float* __restrict__ bias,
                                bf16_t* __restrict__ act,
                                int lc, long long total)
{
    long long t = (long long)blockIdx.x * blockDim.x + threadIdx.x;
    if (t >= total) return;
    int c = (int)(t & ((1 << lc) - 1));
    float v = acc[t] + bias[c];
    v = (v >= 0.0f) ? v : SLOPE * v;
    act[t] = __float2bfloat16(v);
}

// ---------------------------------------------------------------------------
// Final FC: one block per graph, dot over 64*128 = 8192 elems, ReLU
// ---------------------------------------------------------------------------
__global__ void __launch_bounds__(256)
fc_kernel(const bf16_t* __restrict__ act, const float* __restrict__ Wfc,
          const float* __restrict__ bfc, float* __restrict__ out)
{
    __shared__ float red[256];
    int b = blockIdx.x, tid = threadIdx.x;
    const bf16_t* row = act + (size_t)b * (NUM_NODES * H3);
    float s = 0.0f;
    for (int j = tid; j < NUM_NODES * H3; j += 256)
        s += __bfloat162float(row[j]) * Wfc[j];
    red[tid] = s;
    __syncthreads();
    for (int off = 128; off > 0; off >>= 1) {
        if (tid < off) red[tid] += red[tid + off];
        __syncthreads();
    }
    if (tid == 0) out[b] = fmaxf(red[0] + bfc[0], 0.0f);
}

// ---------------------------------------------------------------------------
// Host-side launcher
// ---------------------------------------------------------------------------
static inline int blocks_for(long long total, int bs)
{
    return (int)((total + bs - 1) / bs);
}

extern "C" void kernel_launch(void* const* d_in, const int* in_sizes, int n_in,
                              void* d_out, int out_size, void* d_ws, size_t ws_size,
                              hipStream_t stream)
{
    const float* x   = (const float*)d_in[0];
    const int*   ei  = (const int*)d_in[1];
    const float* W1  = (const float*)d_in[2];
    const float* b1  = (const float*)d_in[3];
    const float* W2  = (const float*)d_in[4];
    const float* b2  = (const float*)d_in[5];
    const float* W3  = (const float*)d_in[6];
    const float* b3  = (const float*)d_in[7];
    const float* Wfc = (const float*)d_in[8];
    const float* bfc = (const float*)d_in[9];
    float* out = (float*)d_out;
    (void)in_sizes; (void)n_in; (void)out_size; (void)ws_size;

    const int* src = ei;
    const int* dst = ei + NEDGE;

    // workspace carve-up (256B aligned)
    char* w = (char*)d_ws;
    auto carve = [&](size_t bytes) {
        void* p = (void*)w;
        w += (bytes + 255) & ~(size_t)255;
        return p;
    };
    float*  dinv = (float*) carve((size_t)NTOT * 4);
    bf16_t* xb   = (bf16_t*)carve((size_t)NTOT * 32 * 2);       // padded K=32
    bf16_t* wt1  = (bf16_t*)carve((size_t)H1 * 32 * 2);
    bf16_t* wt2  = (bf16_t*)carve((size_t)H2 * H1 * 2);
    bf16_t* wt3  = (bf16_t*)carve((size_t)H3 * H2 * 2);
    bf16_t* bufH = (bf16_t*)carve((size_t)NTOT * H2 * 2);       // linear h
    bf16_t* bufA = (bf16_t*)carve((size_t)NTOT * H2 * 2);       // activations
    float*  accb = (float*) carve((size_t)NTOT * H2 * 4);       // fp32 scatter acc

    const int BS = 256;

    // --- degrees / dinv ---
    deg_init_kernel <<<blocks_for(NTOT, BS),  BS, 0, stream>>>(dinv, NTOT);
    deg_acc_kernel  <<<blocks_for(NEDGE, BS), BS, 0, stream>>>(dst, dinv, NEDGE);
    deg_rsqrt_kernel<<<blocks_for(NTOT, BS),  BS, 0, stream>>>(dinv, NTOT);

    // --- conversions ---
    conv_x_kernel <<<blocks_for((long long)NTOT * 32, BS), BS, 0, stream>>>(x, xb);
    conv_wt_kernel<<<blocks_for((long long)H1 * 32, BS), BS, 0, stream>>>(W1, wt1, F_IN, H1, 32);
    conv_wt_kernel<<<blocks_for((long long)H2 * H1, BS), BS, 0, stream>>>(W2, wt2, H1, H2, H1);
    conv_wt_kernel<<<blocks_for((long long)H3 * H2, BS), BS, 0, stream>>>(W3, wt3, H2, H3, H2);

    // --- layer 1: K=32 (padded), Nc=256, lc=8 ---
    {
        dim3 g(NTOT / 64, H1 / 64);
        wmma_gemm_bf16<<<g, 128, 0, stream>>>(xb, wt1, bufH, NTOT, 32, H1);
        long long tn = (long long)NTOT * H1;
        long long te = (long long)NEDGE * H1;
        self_loop_init_kernel<<<blocks_for(tn, BS), BS, 0, stream>>>(bufH, dinv, accb, 8, tn);
        edge_scatter_kernel  <<<blocks_for(te, BS), BS, 0, stream>>>(bufH, dinv, src, dst, accb, 8, te);
        bias_act_kernel      <<<blocks_for(tn, BS), BS, 0, stream>>>(accb, b1, bufA, 8, tn);
    }

    // --- layer 2: K=256, Nc=512, lc=9 ---
    {
        dim3 g(NTOT / 64, H2 / 64);
        wmma_gemm_bf16<<<g, 128, 0, stream>>>(bufA, wt2, bufH, NTOT, H1, H2);
        long long tn = (long long)NTOT * H2;
        long long te = (long long)NEDGE * H2;
        self_loop_init_kernel<<<blocks_for(tn, BS), BS, 0, stream>>>(bufH, dinv, accb, 9, tn);
        edge_scatter_kernel  <<<blocks_for(te, BS), BS, 0, stream>>>(bufH, dinv, src, dst, accb, 9, te);
        bias_act_kernel      <<<blocks_for(tn, BS), BS, 0, stream>>>(accb, b2, bufA, 9, tn);
    }

    // --- layer 3: K=512, Nc=128, lc=7 ---
    {
        dim3 g(NTOT / 64, H3 / 64);
        wmma_gemm_bf16<<<g, 128, 0, stream>>>(bufA, wt3, bufH, NTOT, H2, H3);
        long long tn = (long long)NTOT * H3;
        long long te = (long long)NEDGE * H3;
        self_loop_init_kernel<<<blocks_for(tn, BS), BS, 0, stream>>>(bufH, dinv, accb, 7, tn);
        edge_scatter_kernel  <<<blocks_for(te, BS), BS, 0, stream>>>(bufH, dinv, src, dst, accb, 7, te);
        bias_act_kernel      <<<blocks_for(tn, BS), BS, 0, stream>>>(accb, b3, bufA, 7, tn);
    }

    // --- final FC + ReLU ---
    fc_kernel<<<BATCH, 256, 0, stream>>>(bufA, Wfc, bfc, out);
}
